// MV_LSTM_74217034875115
// MI455X (gfx1250) — matrix operand order repaired and checked by reading, not compiled, not run on hardware
//
#include <hip/hip_runtime.h>

typedef __attribute__((ext_vector_type(16))) _Float16 v16h;
typedef __attribute__((ext_vector_type(8)))  float    v8f;
typedef __attribute__((ext_vector_type(4)))  float    f32x4;
typedef __attribute__((ext_vector_type(2)))  float    f32x2;

#define B_SZ 2048
#define T_SZ 512
#define F_SZ 64
#define H_SZ 20
#define GLN  44      // gates LDS per-source-lane stride (dwords, 16B aligned)
#define HST  32      // h state row stride (padded from 20 -> branchless A rebuild)
#define LDSZ (32 * GLN + 16 * HST + 16)   // gates + h + out partials (dwords)

// gates element (m,n) lives at lane_src*GLN + (n/16)*8 + (m%8),
// lane_src = (n%16) + 16*(m/8)   [C/D-layout source lane, tile n/16 owned by wave n/16]
__device__ __forceinline__ int gidx(int m, int n) {
    return ((n & 15) + 16 * (m >> 3)) * GLN + (n >> 4) * 8 + (m & 7);
}

__device__ __forceinline__ float sigmoid_fast(float v) {
    return 1.0f / (1.0f + __expf(-v));
}
__device__ __forceinline__ float tanh_fast(float v) {
    return 1.0f - 2.0f / (__expf(2.0f * v) + 1.0f);   // saturates correctly
}

// CDNA5 split workgroup barrier (signal / wait on barrier -1)
__device__ __forceinline__ void bar_signal() {
    asm volatile("" ::: "memory");
#if __has_builtin(__builtin_amdgcn_s_barrier_signal)
    __builtin_amdgcn_s_barrier_signal(-1);
#else
    asm volatile("s_barrier_signal -1" ::: "memory");
#endif
}
__device__ __forceinline__ void bar_wait() {
#if __has_builtin(__builtin_amdgcn_s_barrier_wait)
    __builtin_amdgcn_s_barrier_wait(-1);
#else
    asm volatile("s_barrier_wait -1" ::: "memory");
#endif
    asm volatile("" ::: "memory");
}
__device__ __forceinline__ void lds_drain() {
    asm volatile("s_wait_dscnt 0" ::: "memory");
}

// 5 waves (160 threads) cooperate on one 16-batch tile across all T=512 steps.
// Wave w owns gate N-tile w (16 of 80 columns): 3 WMMAs/step. Gate nonlinearity
// is split 2 elements/lane over the whole block. Split barriers are hidden by
// overlapping the next x A-tile fetch and the h-independent input GEMM.
__global__ __launch_bounds__(160, 1)
void mv_lstm_fused_kernel(const float* __restrict__ x,
                          const float* __restrict__ W_ih,
                          const float* __restrict__ W_hh,
                          const float* __restrict__ b_ih,
                          const float* __restrict__ b_hh,
                          const float* __restrict__ W_lin,
                          const float* __restrict__ b_lin,
                          float* __restrict__ out)
{
    __shared__ __align__(16) float smem[LDSZ];
    float* g_lds = smem;                  // [32][GLN] gates, source-lane-major
    float* h_lds = smem + 32 * GLN;       // [16][HST] h state, zero-padded
    float* o_lds = h_lds + 16 * HST;      // [16] output partials

    const int tid   = threadIdx.x;
    const int lane  = tid & 31;
    const int wslot = tid >> 5;           // 0..4 = owned gate N-tile
    const int bbase = blockIdx.x * 16;
    const int col   = lane & 15;          // N column (B/C/D) and A-row M
    const int half  = lane >> 4;          // K-half selector

    // zero padded h state (pad cols 20..31 stay zero forever) and out partials
    for (int i = tid; i < 16 * HST; i += 160) h_lds[i] = 0.0f;
    if (tid < 16) o_lds[tid] = 0.0f;

    // ---- this wave's B-matrices in WMMA B layout (lane = N col, K by half) ----
    const int n = wslot * 16 + col;
    v16h bih0, bih1, bhh;
    v8f  biasC;
    {
        const float bv = b_ih[n] + b_hh[n];
#pragma unroll
        for (int r = 0; r < 8; ++r) biasC[r] = bv;
#pragma unroll
        for (int j = 0; j < 8; ++j) {                      // B[K,n] = W_ih[n,K]
            const int K = half * 16 + 2 * j;
            f32x2 w0 = *(const f32x2*)(W_ih + n * F_SZ + K);
            f32x2 w1 = *(const f32x2*)(W_ih + n * F_SZ + 32 + K);
            bih0[2 * j] = (_Float16)w0.x;  bih0[2 * j + 1] = (_Float16)w0.y;
            bih1[2 * j] = (_Float16)w1.x;  bih1[2 * j + 1] = (_Float16)w1.y;
        }
#pragma unroll
        for (int j = 0; j < 8; ++j) {                      // W_hh, K=20 padded to 32
            const int K  = half * 16 + 2 * j;
            const int Kc = (K < H_SZ) ? K : 18;            // clamp addr, select 0
            f32x2 w = *(const f32x2*)(W_hh + n * H_SZ + Kc);
            bhh[2 * j]     = (_Float16)((K     < H_SZ) ? w.x : 0.0f);
            bhh[2 * j + 1] = (_Float16)((K + 1 < H_SZ) ? w.y : 0.0f);
        }
    }

    // ---- gate-math mapping: 2 elements per lane (320 = 160*2, no divergence) ----
    int aI[2], aF[2], aG[2], aO[2], hw[2], jw[2], mk[2];
    float c_st[2], wacc[2];
#pragma unroll
    for (int k = 0; k < 2; ++k) {
        const int e = tid + 160 * k;
        const int m = e / H_SZ;
        const int j = e - m * H_SZ;
        aI[k] = gidx(m, j);        aF[k] = gidx(m, j + 20);
        aG[k] = gidx(m, j + 40);   aO[k] = gidx(m, j + 60);
        hw[k] = m * HST + j;  jw[k] = j;  mk[k] = m;
        c_st[k] = 0.0f;  wacc[k] = 0.0f;
    }

    const long rowA = (long)(bbase + col) * T_SZ;   // this lane's A-row in x
    float* hrow = h_lds + col * HST;

    // A-tile builder: dwords 0-7: K=16h+2j
    auto buildA = [&](const float* src) -> v16h {
        v16h a;
#pragma unroll
        for (int j = 0; j < 8; ++j) {
            const int kk = 16 * half + 2 * j;
            f32x2 v = *(const f32x2*)(src + kk);
            a[2 * j]     = (_Float16)v.x;
            a[2 * j + 1] = (_Float16)v.y;
        }
        return a;
    };

    __syncthreads();                       // h/o zero visible to all waves

    // prologue: x A-tiles for t=0; open barrier round B
    const float* xr0 = x + rowA * F_SZ;
    v16h a0 = buildA(xr0);
    v16h a1 = buildA(xr0 + 32);
    bar_signal();                          // round B (h(t=-1) "ready": zeros)

    for (int t = 0; t < T_SZ; ++t) {
        // input GEMM (h-independent) overlaps barrier round B
        v8f acc = __builtin_amdgcn_wmma_f32_16x16x32_f16(
            false, a0, false, bih0, (short)0, biasC, false, false);
        acc = __builtin_amdgcn_wmma_f32_16x16x32_f16(
            false, a1, false, bih1, (short)0, acc, false, false);

        bar_wait();                        // h(t-1) stores visible
        v16h ah = buildA(hrow);            // padded rows: branchless
        acc = __builtin_amdgcn_wmma_f32_16x16x32_f16(
            false, ah, false, bhh, (short)0, acc, false, false);

        // own gate tile to LDS: 2x ds_store_b128
        f32x4 lo = { acc[0], acc[1], acc[2], acc[3] };
        f32x4 hi = { acc[4], acc[5], acc[6], acc[7] };
        *(f32x4*)(g_lds + lane * GLN + wslot * 8)     = lo;
        *(f32x4*)(g_lds + lane * GLN + wslot * 8 + 4) = hi;
        lds_drain();
        bar_signal();                      // round A: gates ready

        // overlap: next timestep's x A-tiles + W_lin row + prefetch
        const int tn = (t < T_SZ - 1) ? (t + 1) : t;   // clamp: no OOB on last iter
        const float* xn = x + (rowA + tn) * F_SZ;
        a0 = buildA(xn);
        a1 = buildA(xn + 32);
        __builtin_prefetch(xn + F_SZ, 0, 3);           // WGP-scope global_prefetch_b8
        const float wlv0 = W_lin[t * H_SZ + jw[0]];
        const float wlv1 = W_lin[t * H_SZ + jw[1]];

        bar_wait();                        // all 5 waves' gate tiles visible

        // gate nonlinearity: 2 elements per lane
#pragma unroll
        for (int k = 0; k < 2; ++k) {
            const float i = sigmoid_fast(g_lds[aI[k]]);
            const float f = sigmoid_fast(g_lds[aF[k]]);
            const float g = tanh_fast(g_lds[aG[k]]);
            const float o = sigmoid_fast(g_lds[aO[k]]);
            const float c = f * c_st[k] + i * g;
            c_st[k] = c;
            const float h = o * tanh_fast(c);
            h_lds[hw[k]] = h;                          // next step's A source
            wacc[k] += h * ((k == 0) ? wlv0 : wlv1);   // fused final linear
        }
        lds_drain();
        bar_signal();                      // round B: h(t) ready
    }
    bar_wait();                            // close final round B

    // ---- final reduction: out[b] = sum over (t,j) + b_lin ----
    atomicAdd(&o_lds[mk[0]], wacc[0]);
    atomicAdd(&o_lds[mk[1]], wacc[1]);
    __syncthreads();
    if (tid < 16) out[bbase + tid] = o_lds[tid] + b_lin[0];
}

extern "C" void kernel_launch(void* const* d_in, const int* in_sizes, int n_in,
                              void* d_out, int out_size, void* d_ws, size_t ws_size,
                              hipStream_t stream) {
    (void)in_sizes; (void)n_in; (void)d_ws; (void)ws_size; (void)out_size;
    const float* x     = (const float*)d_in[0];
    const float* W_ih  = (const float*)d_in[1];
    const float* W_hh  = (const float*)d_in[2];
    const float* b_ih  = (const float*)d_in[3];
    const float* b_hh  = (const float*)d_in[4];
    const float* W_lin = (const float*)d_in[5];
    const float* b_lin = (const float*)d_in[6];
    float* out = (float*)d_out;

    // one workgroup (5 waves, 160 threads) per 16-batch tile: 128 blocks, 640 waves
    dim3 grid(B_SZ / 16);
    dim3 block(160);
    mv_lstm_fused_kernel<<<grid, block, 0, stream>>>(x, W_ih, W_hh, b_ih, b_hh,
                                                     W_lin, b_lin, out);
}
